// GroupLinearEncoder_68281390072014
// MI455X (gfx1250) — matrix operand order repaired and checked by reading, not compiled, not run on hardware
//
#include <hip/hip_runtime.h>
#include <hip/hip_bf16.h>

typedef __attribute__((ext_vector_type(16))) __bf16 v16bf;
typedef __attribute__((ext_vector_type(16), aligned(16))) __bf16 v16bfA;
typedef __attribute__((ext_vector_type(8)))  float    v8f;
typedef __attribute__((ext_vector_type(4)))  float    v4f;
typedef __attribute__((ext_vector_type(8)))  unsigned v8u;
typedef __attribute__((ext_vector_type(2)))  unsigned v2u;

// Two f32 -> one dword {bf16(hi):bf16(lo)} via a single v_perm_b32.
__device__ __forceinline__ unsigned pack_bf16x2(float lo, float hi) {
  return __builtin_amdgcn_perm(__builtin_bit_cast(unsigned, hi),
                               __builtin_bit_cast(unsigned, lo), 0x07060302u);
}

// 16-bf16 WMMA operand from two 8-float runs (each 16B-aligned): 4 b128 loads
// + 8 v_perm.
__device__ __forceinline__ v16bf ldpack16(const float* __restrict__ p0,
                                          const float* __restrict__ p1) {
  v4f a = *(const v4f*)p0;
  v4f b = *(const v4f*)(p0 + 4);
  v4f c = *(const v4f*)p1;
  v4f e = *(const v4f*)(p1 + 4);
  v8u u;
  u[0] = pack_bf16x2(a[0], a[1]); u[1] = pack_bf16x2(a[2], a[3]);
  u[2] = pack_bf16x2(b[0], b[1]); u[3] = pack_bf16x2(b[2], b[3]);
  u[4] = pack_bf16x2(c[0], c[1]); u[5] = pack_bf16x2(c[2], c[3]);
  u[6] = pack_bf16x2(e[0], e[1]); u[7] = pack_bf16x2(e[2], e[3]);
  return __builtin_bit_cast(v16bf, u);
}

// ---------------------------------------------------------------------------
// Fused FactorLinear2D partials:
//   hpart[z][n,k] = sum_{c in slice z} sum_l w1[k,l] * x[n,l,c] * w2[k,c]
// Grid: (H/256, N/16, CSPLIT), block 256 (8 waves); wave owns TWO k-tiles.
// x staged in LDS as bf16 in A-operand element order (per (l,n) row:
// [c0..7, c16..23, c8..15, c24..31]) so a lane's A operand is one contiguous
// 32B run at offset hi*32 -> 2x ds_load_b128, zero shuffles.
// ---------------------------------------------------------------------------
template <int L, int C, int CSPLIT>
__global__ __launch_bounds__(256) void feat_proj_wmma(
    const float* __restrict__ x, const float* __restrict__ w1,
    const float* __restrict__ w2, float* __restrict__ hpart) {
  constexpr int H = 1024;
  constexpr int LBLK = 32;
  constexpr int PITCH = 40;  // bf16 elems per row: 32 data + 8 pad (80B rows)
  constexpr int CCHUNK = C / CSPLIT;  // multiple of 32
  __shared__ __bf16 xs[LBLK * 16 * PITCH];

  const int tid  = threadIdx.x;
  const int wave = tid >> 5;
  const int lane = tid & 31;
  const int ln   = lane & 15;
  const int hi   = lane >> 4;
  const int kBase = blockIdx.x * 256 + wave * 32;  // two 16-wide k tiles
  const int nBase = blockIdx.y * 16;
  const int cLo   = blockIdx.z * CCHUNK;
  const int boff  = hi * 8;            // B operand K-run offset per half-wave
  const int kRow0 = kBase + ln;
  const int kRow1 = kBase + 16 + ln;

  v8f acc0 = {}, acc1 = {};
  const v8f czero = {};

  for (int c0 = cLo; c0 < cLo + CCHUNK; c0 += 32) {
    // B tiles = w2[kRow, c0+boff .. +8) ++ w2[kRow, c0+16+boff .. +8), resident across the whole l loop
    const float* b0p = w2 + (size_t)kRow0 * C + c0 + boff;
    const float* b1p = w2 + (size_t)kRow1 * C + c0 + boff;
    v16bf bmat0 = ldpack16(b0p, b0p + 16);
    v16bf bmat1 = ldpack16(b1p, b1p + 16);

    for (int lb = 0; lb < L; lb += LBLK) {
      __syncthreads();  // previous compute done before restaging
#pragma unroll
      for (int t = 0; t < 16; ++t) {
        int id  = tid + t * 256;  // 0..4095 = 16n * 32l * 8(float4)
        int c4  = id & 7;
        int rem = id >> 3;
        int l   = rem & (LBLK - 1);
        int n   = rem >> 5;
        constexpr int cmap[8] = {0, 1, 4, 5, 2, 3, 6, 7};
        v2u* dst = (v2u*)&xs[(l * 16 + n) * PITCH + cmap[c4] * 4];
        int lg = lb + l;
        v2u pk = {0u, 0u};
        if (lg < L) {
          v4f v = *(const v4f*)(x + (size_t)(nBase + n) * L * C +
                                (size_t)lg * C + c0 + c4 * 4);
          pk[0] = pack_bf16x2(v[0], v[1]);
          pk[1] = pack_bf16x2(v[2], v[3]);
        }
        *dst = pk;
      }
      __syncthreads();

      const int lmax = (L - lb < LBLK) ? (L - lb) : LBLK;
#pragma unroll 2
      for (int l = 0; l < lmax; ++l) {
        v16bf amat = __builtin_bit_cast(
            v16bf, *(const v16bfA*)&xs[(l * 16 + ln) * PITCH + hi * 16]);
        v8f d0 = __builtin_amdgcn_wmma_f32_16x16x32_bf16(
            false, amat, false, bmat0, (short)0, czero, false, false);
        v8f d1 = __builtin_amdgcn_wmma_f32_16x16x32_bf16(
            false, amat, false, bmat1, (short)0, czero, false, false);
        const int lg = lb + l;
        float w1a = w1[(size_t)kRow0 * L + lg];  // per-lane (column k) scales
        float w1b = w1[(size_t)kRow1 * L + lg];
#pragma unroll
        for (int r = 0; r < 8; ++r) {
          acc0[r] += w1a * d0[r];
          acc1[r] += w1b * d1[r];
        }
      }
    }
  }
  float* hz = hpart + (size_t)blockIdx.z * 64 * H;
  const int n0 = nBase + hi * 8;
#pragma unroll
  for (int r = 0; r < 8; ++r) {
    hz[(size_t)(n0 + r) * H + kBase + ln]      = acc0[r];
    hz[(size_t)(n0 + r) * H + kBase + 16 + ln] = acc1[r];
  }
}

// ---------------------------------------------------------------------------
// Sum CSPLIT partials, BatchNorm (biased batch stats over N=64) both streams,
// then latent = 0.5*(bn0+bn1). Grid: H blocks of 64 threads (thread = n).
// ---------------------------------------------------------------------------
template <int CSPLIT>
__global__ __launch_bounds__(64) void bn_combine(
    const float* __restrict__ h0, const float* __restrict__ h1,
    const float* __restrict__ g0, const float* __restrict__ b0,
    const float* __restrict__ g1, const float* __restrict__ b1,
    float* __restrict__ latent) {
  constexpr int NB = 64, H = 1024;
  const int k = blockIdx.x;
  const int n = threadIdx.x;
  __shared__ float sh[4][NB];
  float v0 = 0.0f, v1 = 0.0f;
#pragma unroll
  for (int z = 0; z < CSPLIT; ++z) {
    v0 += h0[(size_t)z * NB * H + n * H + k];
    v1 += h1[(size_t)z * NB * H + n * H + k];
  }
  sh[0][n] = v0; sh[1][n] = v0 * v0; sh[2][n] = v1; sh[3][n] = v1 * v1;
  __syncthreads();
  for (int s = NB / 2; s > 0; s >>= 1) {
    if (n < s) {
#pragma unroll
      for (int q = 0; q < 4; ++q) sh[q][n] += sh[q][n + s];
    }
    __syncthreads();
  }
  const float inv = 1.0f / NB;
  float mu0 = sh[0][0] * inv, var0 = sh[1][0] * inv - mu0 * mu0;
  float mu1 = sh[2][0] * inv, var1 = sh[3][0] * inv - mu1 * mu1;
  float y0 = g0[k] * (v0 - mu0) * rsqrtf(var0 + 1e-5f) + b0[k];
  float y1 = g1[k] * (v1 - mu1) * rsqrtf(var1 + 1e-5f) + b1[k];
  latent[n * H + k] = 0.5f * (y0 + y1);
}

// ---------------------------------------------------------------------------
// M=64 GEMM: C[m,j] = sum_k A[m,k]*B[j,k] (+ biasScale*bias[j]).
// A: 64xK f32 row-major, B: NxK f32 row-major, K multiple of 32.
// Grid: (ceil(N/256)), block 256. Each wave owns TWO j-tiles x ALL FOUR
// m-tiles (8 accumulators); one packed B pair is reused across 4 m-tiles, so
// B (the big operand, e.g. embed_w 324MB) is streamed from HBM exactly once.
// ---------------------------------------------------------------------------
__global__ __launch_bounds__(256) void gemm_nt_m64_wmma(
    const float* __restrict__ A, const float* __restrict__ B,
    const float* __restrict__ bias, float biasScale, float* __restrict__ Cout,
    int N, int K) {
  const int tid  = threadIdx.x;
  const int wave = tid >> 5;
  const int lane = tid & 31;
  const int ln = lane & 15, hi = lane >> 4;
  const int jBase = blockIdx.x * 256 + wave * 32;
  if (jBase >= N) return;  // wave-uniform
  const int aoff = hi * 16;
  const int boff = hi * 16;
  int bj0 = jBase + ln;      if (bj0 > N - 1) bj0 = N - 1;  // clamp edge rows
  int bj1 = jBase + 16 + ln; if (bj1 > N - 1) bj1 = N - 1;

  const float* brow0 = B + (size_t)bj0 * K;
  const float* brow1 = B + (size_t)bj1 * K;
  const float* arow[4];
#pragma unroll
  for (int mt = 0; mt < 4; ++mt) arow[mt] = A + (size_t)(mt * 16 + ln) * K;

  v8f acc[4][2] = {};
  for (int k0 = 0; k0 < K; k0 += 32) {
    __builtin_prefetch(brow0 + k0 + 256, 0, 1);  // global_prefetch_b8
    __builtin_prefetch(brow1 + k0 + 256, 0, 1);
    v16bf bm0 = ldpack16(brow0 + k0 + boff, brow0 + k0 + boff + 8);
    v16bf bm1 = ldpack16(brow1 + k0 + boff, brow1 + k0 + boff + 8);
#pragma unroll
    for (int mt = 0; mt < 4; ++mt) {
      v16bf am = ldpack16(arow[mt] + k0 + aoff, arow[mt] + k0 + aoff + 8);
      acc[mt][0] = __builtin_amdgcn_wmma_f32_16x16x32_bf16(
          false, am, false, bm0, (short)0, acc[mt][0], false, false);
      acc[mt][1] = __builtin_amdgcn_wmma_f32_16x16x32_bf16(
          false, am, false, bm1, (short)0, acc[mt][1], false, false);
    }
  }
  const int j0 = jBase + ln;
  const int j1 = jBase + 16 + ln;
  const float bv0 = (bias && j0 < N) ? biasScale * bias[j0] : 0.0f;
  const float bv1 = (bias && j1 < N) ? biasScale * bias[j1] : 0.0f;
#pragma unroll
  for (int mt = 0; mt < 4; ++mt) {
    const int m0 = mt * 16 + hi * 8;
    if (j0 < N) {
#pragma unroll
      for (int r = 0; r < 8; ++r)
        Cout[(size_t)(m0 + r) * N + j0] = acc[mt][0][r] + bv0;
    }
    if (j1 < N) {
#pragma unroll
      for (int r = 0; r < 8; ++r)
        Cout[(size_t)(m0 + r) * N + j1] = acc[mt][1][r] + bv1;
    }
  }
}

// ---------------------------------------------------------------------------
// u[n,p] = t[n,p] + s_all[n, idx[n]*2048 + p] + fc_b[idx[n]*2048 + p]
// ---------------------------------------------------------------------------
__global__ __launch_bounds__(256) void gather_combine(
    const float* __restrict__ t, const float* __restrict__ s_all,
    const float* __restrict__ fc_b, const int* __restrict__ idx,
    float* __restrict__ u) {
  int i = blockIdx.x * 256 + threadIdx.x;
  if (i >= 64 * 2048) return;
  int n = i >> 11;
  int p = i & 2047;
  int g = idx[n];
  u[i] = t[i] + s_all[(size_t)n * 16384 + g * 2048 + p] + fc_b[g * 2048 + p];
}

extern "C" void kernel_launch(void* const* d_in, const int* in_sizes, int n_in,
                              void* d_out, int out_size, void* d_ws, size_t ws_size,
                              hipStream_t stream) {
  (void)in_sizes; (void)n_in; (void)out_size; (void)ws_size;
  const float* x0       = (const float*)d_in[0];   // (64,257,1024)
  const float* x1       = (const float*)d_in[1];   // (64,197,768)
  const float* w1_0     = (const float*)d_in[2];   // (1024,257)
  const float* w2_0     = (const float*)d_in[3];   // (1024,1024)
  const float* gamma0   = (const float*)d_in[4];
  const float* beta0    = (const float*)d_in[5];
  const float* w1_1     = (const float*)d_in[6];   // (1024,197)
  const float* w2_1     = (const float*)d_in[7];   // (1024,768)
  const float* gamma1   = (const float*)d_in[8];
  const float* beta1    = (const float*)d_in[9];
  const float* shared_w = (const float*)d_in[10];  // (2048,1024)
  const float* fc_w     = (const float*)d_in[11];  // (16384,1024)
  const float* fc_b     = (const float*)d_in[12];  // (16384,)
  const float* embed_w  = (const float*)d_in[13];  // (39548,2048)
  const float* embed_b  = (const float*)d_in[14];  // (39548,)
  const int*   indices  = (const int*)d_in[15];    // (64,)
  float* out = (float*)d_out;                      // (64,39548)

  constexpr int CSPLIT = 4;
  float* ws     = (float*)d_ws;
  float* hraw0  = ws;                          // CSPLIT * 64*1024 partials
  float* hraw1  = hraw0 + CSPLIT * 64 * 1024;  // CSPLIT * 64*1024 partials
  float* latent = hraw1 + CSPLIT * 64 * 1024;  // 64*1024
  float* t      = latent + 64 * 1024;          // 64*2048
  float* s_all  = t + 64 * 2048;               // 64*16384 (all 8 groups)
  float* u      = s_all + 64 * 16384;          // 64*2048

  // Stage 1: fused FactorLinear2D partials (bf16 WMMA, f32 accum), c-split
  // across grid.z for occupancy on the FLOP-heavy stage.
  feat_proj_wmma<257, 1024, CSPLIT>
      <<<dim3(4, 4, CSPLIT), 256, 0, stream>>>(x0, w1_0, w2_0, hraw0);
  feat_proj_wmma<197, 768, CSPLIT>
      <<<dim3(4, 4, CSPLIT), 256, 0, stream>>>(x1, w1_1, w2_1, hraw1);

  // Stage 2: sum partials + batchnorm (batch stats) + mean of the two streams
  bn_combine<CSPLIT><<<1024, 64, 0, stream>>>(hraw0, hraw1, gamma0, beta0,
                                              gamma1, beta1, latent);

  // Stage 3: t = latent @ shared_w^T ; s_all = latent @ fc_w^T (all 8 groups)
  gemm_nt_m64_wmma<<<8, 256, 0, stream>>>(latent, shared_w, nullptr, 0.0f,
                                          t, 2048, 1024);
  gemm_nt_m64_wmma<<<64, 256, 0, stream>>>(latent, fc_w, nullptr, 0.0f,
                                           s_all, 16384, 1024);

  // Stage 4: per-sample group gather + combine projections
  gather_combine<<<(64 * 2048 + 255) / 256, 256, 0, stream>>>(t, s_all, fc_b, indices, u);

  // Stage 5: single fused output GEMM, embed_w streamed exactly once:
  // out = u @ embed_w^T + 2*embed_b
  gemm_nt_m64_wmma<<<(39548 + 255) / 256, 256, 0, stream>>>(
      u, embed_w, embed_b, 2.0f, out, 39548, 2048);
}
